// MotionLSTM_36816459661412
// MI455X (gfx1250) — matrix operand order repaired and checked by reading, not compiled, not run on hardware
//
#include <hip/hip_runtime.h>
#include <cstdint>

typedef __bf16 bf16_t;
typedef __attribute__((ext_vector_type(16))) __bf16 v16bf;
typedef __attribute__((ext_vector_type(8)))  float  v8f;

struct Frag {
    union { uint4 u[2]; v16bf v; };
};

__device__ __forceinline__ float sigm_f(float x) {
    return 1.0f / (1.0f + __expf(-x));
}
__device__ __forceinline__ float tanh_f(float x) {
    // tanh(x) = 2*sigmoid(2x) - 1
    return 2.0f / (1.0f + __expf(-2.0f * x)) - 1.0f;
}

// ---------------------------------------------------------------------------
// Pack f32 [rows][kin] -> bf16 [rows][kout] (zero pad K)
// ---------------------------------------------------------------------------
__global__ void pack_pad_bf16(const float* __restrict__ in, bf16_t* __restrict__ out,
                              int rows, int kin, int kout) {
    int idx = blockIdx.x * blockDim.x + threadIdx.x;
    if (idx >= rows * kout) return;
    int r = idx / kout;
    int k = idx - r * kout;
    float v = (k < kin) ? in[(size_t)r * kin + k] : 0.0f;
    out[idx] = (bf16_t)v;
}

// ---------------------------------------------------------------------------
// C[M][N] = A[M][K] * W[N][K]^T + bias0 + bias1   (bf16 in, bf16 out, f32 acc)
// block = 256 threads = 8 waves; block tile 128(M) x 64(N); wave: 16x64
// grid = (N/64, M/128)
// ---------------------------------------------------------------------------
__global__ __launch_bounds__(256) void gemm_bias_bf16(
    const bf16_t* __restrict__ A, const bf16_t* __restrict__ W,
    const float* __restrict__ bias0, const float* __restrict__ bias1,
    bf16_t* __restrict__ C, int M, int N, int K)
{
    const int tid   = threadIdx.x;
    const int w     = tid >> 5;
    const int lane  = tid & 31;
    const int lhalf = lane >> 4;   // 0 or 1
    const int l16   = lane & 15;

    const int mbase = blockIdx.y * 128 + w * 16;
    const int nbase = blockIdx.x * 64;

    v8f acc[4] = {};

    const bf16_t* arow = A + (size_t)(mbase + l16) * K;
    for (int k0 = 0; k0 < K; k0 += 32) {
        // A fragment (16x32 bf16): lane<16 -> K {0..7,16..23}, lane>=16 -> {8..15,24..31}
        Frag a;
        const int oa = k0 + lhalf * 8;
        a.u[0] = *(const uint4*)(arow + oa);
        a.u[1] = *(const uint4*)(arow + oa + 16);
#pragma unroll
        for (int nt = 0; nt < 4; ++nt) {
            // B fragment (32x16): lane<16 -> K {0..7,16..23}, lane>=16 -> {8..15,24..31}
            Frag b;
            const bf16_t* wrow = W + (size_t)(nbase + nt * 16 + l16) * K;
            const int ob = k0 + lhalf * 8;
            b.u[0] = *(const uint4*)(wrow + ob);
            b.u[1] = *(const uint4*)(wrow + ob + 16);
            acc[nt] = __builtin_amdgcn_wmma_f32_16x16x32_bf16(
                false, a.v, false, b.v, (short)0, acc[nt], false, false);
        }
    }

#pragma unroll
    for (int nt = 0; nt < 4; ++nt) {
        const int col = nbase + nt * 16 + l16;
        const float bv = bias0[col] + bias1[col];
#pragma unroll
        for (int r = 0; r < 8; ++r) {
            const int row = mbase + r + lhalf * 8;   // C layout: lanes16-31 -> M=r+8
            C[(size_t)row * N + col] = (bf16_t)(acc[nt][r] + bv);
        }
    }
}

// ---------------------------------------------------------------------------
// LSTM scan over T. One block = 16 batch rows, 8 waves.
// Wave w owns hidden cols 16w..16w+15 and gate cols 16w + {0,128,256,384},
// so the cell update is wave-local. W_hh^T fragments persist in VGPRs.
// The 16x512 bf16 gate tile for step t+1 is prefetched into a double-buffered
// LDS tile with global_load_async_to_lds_b128 while step t's WMMAs run;
// completion gated with s_wait_asynccnt.
// grid = B/16, block = 256
// ---------------------------------------------------------------------------
__global__ __launch_bounds__(256) void lstm_scan(
    const bf16_t* __restrict__ whh,    // [512][128] bf16
    const bf16_t* __restrict__ xproj,  // [B*T][512] bf16 (x@Wih^T + biases)
    bf16_t* __restrict__ hseq,         // [B*T][128] bf16 (if write_seq)
    float*  __restrict__ hlast,        // [B][128] f32 (if non-null)
    int T, int write_seq)
{
    __shared__ __align__(16) bf16_t h_lds[16 * 128];
    __shared__ __align__(16) bf16_t xtile[2][16 * 512];   // double-buffered gate tile

    const int tid   = threadIdx.x;
    const int w     = tid >> 5;
    const int lane  = tid & 31;
    const int lhalf = lane >> 4;
    const int l16   = lane & 15;
    const int bb0   = blockIdx.x * 16;
    const int jcol  = w * 16 + l16;      // hidden col this lane produces

    // Persistent B fragments of Whh^T: [gate i/f/g/o][k-step], 16 x v16bf.
    Frag bfrag[4][4];
#pragma unroll
    for (int g = 0; g < 4; ++g) {
        const bf16_t* wrow = whh + (size_t)(g * 128 + w * 16 + l16) * 128;
#pragma unroll
        for (int kk = 0; kk < 4; ++kk) {
            const int ob = kk * 32 + lhalf * 8;
            bfrag[g][kk].u[0] = *(const uint4*)(wrow + ob);
            bfrag[g][kk].u[1] = *(const uint4*)(wrow + ob + 16);
        }
    }

    // Async-prefetch plumbing: tile = 16 rows x 1KB = 1024 chunks of 16B;
    // each of the 256 threads owns 4 chunks.
    const bf16_t* gsrc[4];
    unsigned ldsoff[2][4];
#pragma unroll
    for (int q = 0; q < 4; ++q) {
        const int cidx = q * 256 + tid;   // chunk index 0..1023
        const int brow = cidx >> 6;       // 64 chunks per batch row
        const int cc   = cidx & 63;
        gsrc[q] = xproj + ((size_t)(bb0 + brow) * T) * 512 + cc * 8;
        ldsoff[0][q] = (unsigned)(size_t)&xtile[0][cidx * 8];
        ldsoff[1][q] = (unsigned)(size_t)&xtile[1][cidx * 8];
    }

    // prefetch t = 0 into buffer 0
#pragma unroll
    for (int q = 0; q < 4; ++q)
        asm volatile("global_load_async_to_lds_b128 %0, %1, off"
                     :: "v"(ldsoff[0][q]), "v"(gsrc[q]) : "memory");

    // h = 0, c = 0
    for (int i = tid; i < 16 * 128; i += 256) h_lds[i] = (bf16_t)0.0f;
    v8f c = {};
    __syncthreads();

    for (int t = 0; t < T; ++t) {
        const int cur = t & 1;
        const bool more = (t + 1 < T);

        // issue next step's tile prefetch before the matrix work
        if (more) {
            const int nxt = cur ^ 1;
#pragma unroll
            for (int q = 0; q < 4; ++q)
                asm volatile("global_load_async_to_lds_b128 %0, %1, off"
                             :: "v"(ldsoff[nxt][q]),
                                "v"(gsrc[q] + (size_t)(t + 1) * 512)
                             : "memory");
        }

        v8f ai = {}, af = {}, ag = {}, ao = {};
#pragma unroll
        for (int kk = 0; kk < 4; ++kk) {
            // A fragment from LDS h (16 rows x 32 K)
            Frag a;
            const int oa = kk * 32 + lhalf * 8;
            a.u[0] = *(const uint4*)&h_lds[l16 * 128 + oa];
            a.u[1] = *(const uint4*)&h_lds[l16 * 128 + oa + 16];
            ai = __builtin_amdgcn_wmma_f32_16x16x32_bf16(false, a.v, false, bfrag[0][kk].v, (short)0, ai, false, false);
            af = __builtin_amdgcn_wmma_f32_16x16x32_bf16(false, a.v, false, bfrag[1][kk].v, (short)0, af, false, false);
            ag = __builtin_amdgcn_wmma_f32_16x16x32_bf16(false, a.v, false, bfrag[2][kk].v, (short)0, ag, false, false);
            ao = __builtin_amdgcn_wmma_f32_16x16x32_bf16(false, a.v, false, bfrag[3][kk].v, (short)0, ao, false, false);
        }

        // current tile must have landed; next step's 4 loads may stay in flight
        if (more) asm volatile("s_wait_asynccnt 0x4" ::: "memory");
        else      asm volatile("s_wait_asynccnt 0x0" ::: "memory");
        __syncthreads();   // h_lds reads done + everyone's async tile visible

        const bf16_t* xt = xtile[cur];
        float hnew[8];
#pragma unroll
        for (int r = 0; r < 8; ++r) {
            const int brow = r + lhalf * 8;
            const int xb = brow * 512 + jcol;
            const float iv = sigm_f(ai[r] + (float)xt[xb]);
            const float fv = sigm_f(af[r] + (float)xt[xb + 128]);
            const float gv = tanh_f(ag[r] + (float)xt[xb + 256]);
            const float ov = sigm_f(ao[r] + (float)xt[xb + 384]);
            const float cv = fv * c[r] + iv * gv;
            c[r] = cv;
            hnew[r] = ov * tanh_f(cv);
        }
#pragma unroll
        for (int r = 0; r < 8; ++r) {
            const int brow = r + lhalf * 8;
            h_lds[brow * 128 + jcol] = (bf16_t)hnew[r];
        }
        if (write_seq) {
#pragma unroll
            for (int r = 0; r < 8; ++r) {
                const int brow = r + lhalf * 8;
                hseq[((size_t)(bb0 + brow) * T + t) * 128 + jcol] = (bf16_t)hnew[r];
            }
        }
        __syncthreads();   // h_t visible before next step's reads
    }

    if (hlast != nullptr) {
        for (int i = tid; i < 16 * 128; i += 256)
            hlast[(size_t)bb0 * 128 + i] = (float)h_lds[i];
    }
}

// ---------------------------------------------------------------------------
// FC head: out = sigmoid(relu(h @ Wfc1^T + b1) @ Wfc2^T + b2), per batch row.
// grid = B, block = 64
// ---------------------------------------------------------------------------
__global__ __launch_bounds__(64) void fc_head(
    const float* __restrict__ hlast,  // [B][128]
    const float* __restrict__ wfc1,   // [64][128]
    const float* __restrict__ bfc1,   // [64]
    const float* __restrict__ wfc2,   // [1][64]
    const float* __restrict__ bfc2,   // [1]
    float* __restrict__ out)          // [B]
{
    __shared__ float s[64];
    const int b = blockIdx.x;
    const int j = threadIdx.x;
    const float* h  = hlast + (size_t)b * 128;
    const float* wr = wfc1 + (size_t)j * 128;
    float acc = bfc1[j];
#pragma unroll 4
    for (int k = 0; k < 128; ++k) acc += h[k] * wr[k];
    s[j] = fmaxf(acc, 0.0f) * wfc2[j];
    __syncthreads();
    if (j == 0) {
        float r = bfc2[0];
#pragma unroll 4
        for (int k = 0; k < 64; ++k) r += s[k];
        out[b] = 1.0f / (1.0f + __expf(-r));
    }
}

// ---------------------------------------------------------------------------

extern "C" void kernel_launch(void* const* d_in, const int* in_sizes, int n_in,
                              void* d_out, int out_size, void* d_ws, size_t ws_size,
                              hipStream_t stream) {
    (void)in_sizes; (void)n_in; (void)out_size; (void)ws_size;

    const float* x    = (const float*)d_in[0];   // [512][512][18]
    const float* Wih0 = (const float*)d_in[1];   // [512][18]
    const float* Whh0 = (const float*)d_in[2];   // [512][128]
    const float* bih0 = (const float*)d_in[3];
    const float* bhh0 = (const float*)d_in[4];
    const float* Wih1 = (const float*)d_in[5];   // [512][128]
    const float* Whh1 = (const float*)d_in[6];   // [512][128]
    const float* bih1 = (const float*)d_in[7];
    const float* bhh1 = (const float*)d_in[8];
    const float* Wfc1 = (const float*)d_in[9];   // [64][128]
    const float* bfc1 = (const float*)d_in[10];
    const float* Wfc2 = (const float*)d_in[11];  // [1][64]
    const float* bfc2 = (const float*)d_in[12];

    const int B = 512, T = 512, H = 128, G = 4 * H;  // G = 512 gate cols
    const int M = B * T;                              // 262144 time-parallel rows

    // workspace layout (all chunk sizes multiples of 256 B)
    char* ws = (char*)d_ws;
    size_t off = 0;
    bf16_t* xpad  = (bf16_t*)(ws + off); off += (size_t)M * 32 * 2;   // x padded K 18->32
    bf16_t* w0pad = (bf16_t*)(ws + off); off += (size_t)G * 32 * 2;   // Wih0 padded
    bf16_t* whh0b = (bf16_t*)(ws + off); off += (size_t)G * H * 2;
    bf16_t* wih1b = (bf16_t*)(ws + off); off += (size_t)G * H * 2;
    bf16_t* whh1b = (bf16_t*)(ws + off); off += (size_t)G * H * 2;
    bf16_t* xproj = (bf16_t*)(ws + off); off += (size_t)M * G * 2;    // reused both layers
    bf16_t* h1seq = (bf16_t*)(ws + off); off += (size_t)M * H * 2;
    float*  h2last= (float*)(ws + off);  off += (size_t)B * H * 4;

    // 1) pack / convert to bf16
    {
        int n = M * 32;
        pack_pad_bf16<<<(n + 255) / 256, 256, 0, stream>>>(x, xpad, M, 18, 32);
        n = G * 32;
        pack_pad_bf16<<<(n + 255) / 256, 256, 0, stream>>>(Wih0, w0pad, G, 18, 32);
        n = G * H;
        pack_pad_bf16<<<(n + 255) / 256, 256, 0, stream>>>(Whh0, whh0b, G, H, H);
        pack_pad_bf16<<<(n + 255) / 256, 256, 0, stream>>>(Wih1, wih1b, G, H, H);
        pack_pad_bf16<<<(n + 255) / 256, 256, 0, stream>>>(Whh1, whh1b, G, H, H);
    }

    dim3 ggrid(G / 64, M / 128);

    // 2) layer-0 input projection: xproj = xpad @ w0pad^T + bih0 + bhh0
    gemm_bias_bf16<<<ggrid, 256, 0, stream>>>(xpad, w0pad, bih0, bhh0, xproj, M, G, 32);

    // 3) layer-0 recurrent scan -> h1seq
    lstm_scan<<<B / 16, 256, 0, stream>>>(whh0b, xproj, h1seq, nullptr, T, 1);

    // 4) layer-1 input projection: xproj = h1seq @ wih1^T + bih1 + bhh1
    gemm_bias_bf16<<<ggrid, 256, 0, stream>>>(h1seq, wih1b, bih1, bhh1, xproj, M, G, H);

    // 5) layer-1 recurrent scan -> h2last only
    lstm_scan<<<B / 16, 256, 0, stream>>>(whh1b, xproj, nullptr, h2last, T, 0);

    // 6) FC head -> sigmoid output [B]
    fc_head<<<B, 64, 0, stream>>>(h2last, Wfc1, bfc1, Wfc2, bfc2, (float*)d_out);
}
